// MambaBlock_33861522162381
// MI455X (gfx1250) — compile-verified, matrix operand order checked by reading, not run on hardware
//
#include <hip/hip_runtime.h>
#include <hip/hip_bf16.h>

// ---------------------------------------------------------------------------
// Mamba block for MI455X (gfx1250, wave32, WMMA).
// Pipeline: LN -> in_proj (WMMA bf16) -> causal dw-conv+SiLU -> x_proj ->
//           dt_proj (WMMA bf16, softplus) -> chunked parallel selective scan
//           (3 phases) -> out_proj (WMMA bf16, +residual).
// ---------------------------------------------------------------------------

#define D_MODEL 1024
#define D_STATE 16
#define D_CONV  4
#define D_INNER 2048
#define B_SZ    2
#define L_SEQ   1024
#define BL      (B_SZ * L_SEQ)          // 2048 rows
#define NCHUNK  16
#define CHUNK   (L_SEQ / NCHUNK)        // 64 steps per chunk

typedef __attribute__((ext_vector_type(16))) __bf16 v16bf;
typedef __attribute__((ext_vector_type(8)))  float  v8f;

// ---- helpers ---------------------------------------------------------------

__device__ __forceinline__ unsigned short f2bf(float f) {
    union { float f; unsigned int u; } c; c.f = f;
    unsigned int u = c.u;
    unsigned int lsb = (u >> 16) & 1u;
    u += 0x7fffu + lsb;                 // round-to-nearest-even
    return (unsigned short)(u >> 16);
}

__device__ __forceinline__ float sigmoid_(float x) { return 1.0f / (1.0f + __expf(-x)); }
__device__ __forceinline__ float silu_(float x)    { return x * sigmoid_(x); }
__device__ __forceinline__ float softplus_(float x){ return (x > 20.0f) ? x : log1pf(__expf(x)); }

// Fragment: 16 bf16 per lane (8 VGPRs), ISA layout for 16-bit A-matrix 16x32.
struct Frag {
    union { uint4 q[2]; v16bf v; };
};

// p points at row start + k0 (row-major, K contiguous). half = lane>>4.
// lanes 0-15  : K = k0+0..7  (q0), k0+16..23 (q1)
// lanes 16-31 : K = k0+8..15 (q0), k0+24..31 (q1)
__device__ __forceinline__ Frag load_frag(const unsigned short* p, int half) {
    Frag f;
    f.q[0] = *reinterpret_cast<const uint4*>(p + half * 8);
    f.q[1] = *reinterpret_cast<const uint4*>(p + 16 + half * 8);
    return f;
}

// ---- f32 -> bf16 convert ---------------------------------------------------

__global__ __launch_bounds__(256)
void k_f32_to_bf16(const float* __restrict__ src, unsigned short* __restrict__ dst, int n) {
    int i = blockIdx.x * 256 + threadIdx.x;
    if (i < n) dst[i] = f2bf(src[i]);
}

// ---- LayerNorm (one block per row of 1024) --------------------------------

__global__ __launch_bounds__(256)
void k_layernorm_bf16(const float* __restrict__ x, const float* __restrict__ w,
                      const float* __restrict__ b, unsigned short* __restrict__ out) {
    const int row = blockIdx.x;
    const float* xr = x + (size_t)row * D_MODEL;
    __shared__ float red[256];

    float s = 0.f;
    for (int i = threadIdx.x; i < D_MODEL; i += 256) s += xr[i];
    red[threadIdx.x] = s; __syncthreads();
    for (int st = 128; st > 0; st >>= 1) {
        if (threadIdx.x < st) red[threadIdx.x] += red[threadIdx.x + st];
        __syncthreads();
    }
    const float mu = red[0] * (1.0f / D_MODEL);
    __syncthreads();

    float v = 0.f;
    for (int i = threadIdx.x; i < D_MODEL; i += 256) { float d = xr[i] - mu; v += d * d; }
    red[threadIdx.x] = v; __syncthreads();
    for (int st = 128; st > 0; st >>= 1) {
        if (threadIdx.x < st) red[threadIdx.x] += red[threadIdx.x + st];
        __syncthreads();
    }
    const float rstd = rsqrtf(red[0] * (1.0f / D_MODEL) + 1e-5f);

    for (int i = threadIdx.x; i < D_MODEL; i += 256)
        out[(size_t)row * D_MODEL + i] = f2bf((xr[i] - mu) * rstd * w[i] + b[i]);
}

// ---- WMMA GEMM: D[M,N] = A[M,K](bf16) * W[N,K](bf16)^T + bias, epilogue ----
// MODE 0: +bias            (in_proj -> xz)
// MODE 1: softplus(+bias)  (dt_proj -> delta)
// MODE 2: +bias +residual  (out_proj -> d_out)
// Block: 256 threads = 8 waves in 2x4 grid; wave tile 32x32; block tile 64x128.

template <int MODE>
__global__ __launch_bounds__(256)
void k_gemm_wmma_bf16(const unsigned short* __restrict__ A,
                      const unsigned short* __restrict__ W,
                      const float* __restrict__ bias,
                      const float* __restrict__ resid,
                      float* __restrict__ D, int M, int N, int K) {
    const int lane = threadIdx.x & 31;
    const int wave = threadIdx.x >> 5;
    const int wm = wave >> 2;           // 0..1
    const int wn = wave & 3;            // 0..3
    const int row0 = blockIdx.y * 64  + wm * 32;
    const int col0 = blockIdx.x * 128 + wn * 32;
    const int r    = lane & 15;
    const int half = lane >> 4;

    v8f acc00 = {0,0,0,0,0,0,0,0};
    v8f acc01 = acc00, acc10 = acc00, acc11 = acc00;

    const unsigned short* Ar0 = A + (size_t)(row0 + r)      * K;
    const unsigned short* Ar1 = A + (size_t)(row0 + 16 + r) * K;
    const unsigned short* Wc0 = W + (size_t)(col0 + r)      * K;
    const unsigned short* Wc1 = W + (size_t)(col0 + 16 + r) * K;

    for (int k0 = 0; k0 < K; k0 += 32) {
        Frag a0 = load_frag(Ar0 + k0, half);
        Frag a1 = load_frag(Ar1 + k0, half);
        Frag b0 = load_frag(Wc0 + k0, half);
        Frag b1 = load_frag(Wc1 + k0, half);
        acc00 = __builtin_amdgcn_wmma_f32_16x16x32_bf16(false, a0.v, false, b0.v, (short)0, acc00, false, false);
        acc01 = __builtin_amdgcn_wmma_f32_16x16x32_bf16(false, a0.v, false, b1.v, (short)0, acc01, false, false);
        acc10 = __builtin_amdgcn_wmma_f32_16x16x32_bf16(false, a1.v, false, b0.v, (short)0, acc10, false, false);
        acc11 = __builtin_amdgcn_wmma_f32_16x16x32_bf16(false, a1.v, false, b1.v, (short)0, acc11, false, false);
    }

    // C/D layout: VGPR v: lanes 0-15 -> (M=v, N=lane); lanes 16-31 -> (M=v+8, N=lane-16)
    auto store_tile = [&](v8f acc, int mbase, int nbase) {
        const int col = nbase + r;
        const float bv = bias[col];
#pragma unroll
        for (int v = 0; v < 8; ++v) {
            const int row = mbase + v + half * 8;
            float val = acc[v] + bv;
            if (MODE == 1) val = softplus_(val);
            if (MODE == 2) val += resid[(size_t)row * N + col];
            D[(size_t)row * N + col] = val;
        }
    };
    store_tile(acc00, row0,      col0);
    store_tile(acc01, row0,      col0 + 16);
    store_tile(acc10, row0 + 16, col0);
    store_tile(acc11, row0 + 16, col0 + 16);
}

// ---- depthwise causal conv1d + SiLU ---------------------------------------
// xz: [BL, 2*D_INNER]; xc = cols [0,2048); writes u as f32 and bf16.

__global__ __launch_bounds__(256)
void k_conv_silu(const float* __restrict__ xz, const float* __restrict__ cw,
                 const float* __restrict__ cb, float* __restrict__ u_f,
                 unsigned short* __restrict__ u_b) {
    const int idx = blockIdx.x * 256 + threadIdx.x;   // B*L*D_INNER threads
    const int d  = idx & (D_INNER - 1);
    const int l  = (idx >> 11) & (L_SEQ - 1);
    const int bl = idx >> 11;                         // b*L + l
    float acc = cb[d];
#pragma unroll
    for (int j = 0; j < D_CONV; ++j) {
        const int li = l - (D_CONV - 1) + j;
        if (li >= 0)
            acc += cw[d * D_CONV + j] * xz[(size_t)(bl - l + li) * (2 * D_INNER) + d];
    }
    const float u = silu_(acc);
    u_f[idx] = u;
    u_b[idx] = f2bf(u);
}

// ---- x_proj: BC[BL,32] = u[BL,2048] @ x_proj_w[32,2048]^T + bias ----------
// N=32 is too narrow for a 128-wide WMMA tile; LDS-staged dot products.

__global__ __launch_bounds__(256)
void k_xproj(const float* __restrict__ u, const float* __restrict__ wp,
             const float* __restrict__ bp, float* __restrict__ bc) {
    const int row = blockIdx.x;
    __shared__ float su[D_INNER];
    __shared__ float red[256];
    const float* ur = u + (size_t)row * D_INNER;
    for (int i = threadIdx.x; i < D_INNER; i += 256) su[i] = ur[i];
    __syncthreads();

    const int n   = threadIdx.x >> 3;   // 0..31
    const int sub = threadIdx.x & 7;
    const float* wr = wp + (size_t)n * D_INNER;
    float s = 0.f;
    for (int i = sub; i < D_INNER; i += 8) s += wr[i] * su[i];
    red[threadIdx.x] = s; __syncthreads();
    if (sub == 0) {
        float t = 0.f;
#pragma unroll
        for (int j = 0; j < 8; ++j) t += red[(n << 3) + j];
        bc[(size_t)row * (2 * D_STATE) + n] = t + bp[n];
    }
}

// ---- chunked parallel selective scan ---------------------------------------
// Recurrence h_l = dA_l*h_{l-1} + dBu_l is affine per (d,n): compose per chunk.
// Pass 1 : thread per (b,d,chunk) -> (P[n], S[n]) with h_end = P*h_start + S.
// Mid    : thread per (b,d)       -> prefix over 16 chunks -> hstart per chunk.
// Pass 2 : thread per (b,d,chunk) -> replay 64 steps from hstart, emit
//          y = (sum_n h*C + D*u) * silu(z) as bf16.
// P/S/hstart layout: [b][chunk][n][d]  (d contiguous -> coalesced).

__global__ __launch_bounds__(256)
void k_scan_pass1(const float* __restrict__ delta, const float* __restrict__ u,
                  const float* __restrict__ bc, const float* __restrict__ A_log,
                  float* __restrict__ P, float* __restrict__ S) {
    const int idx = blockIdx.x * 256 + threadIdx.x;   // B*NCHUNK*D_INNER threads
    const int d = idx & (D_INNER - 1);
    const int c = (idx >> 11) & (NCHUNK - 1);
    const int b = idx >> 15;

    float a[D_STATE], Pr[D_STATE], Sr[D_STATE];
#pragma unroll
    for (int n = 0; n < D_STATE; ++n) {
        a[n]  = -__expf(A_log[d * D_STATE + n]);
        Pr[n] = 1.f;
        Sr[n] = 0.f;
    }
    const size_t l0 = (size_t)b * L_SEQ + (size_t)c * CHUNK;
    for (int t = 0; t < CHUNK; ++t) {
        const size_t rl = l0 + t;
        const float dt = delta[rl * D_INNER + d];
        const float uv = u[rl * D_INNER + d];
        const float du = dt * uv;
        const float* bcr = bc + rl * (2 * D_STATE);
#pragma unroll
        for (int n = 0; n < D_STATE; ++n) {
            const float dA = __expf(dt * a[n]);
            Sr[n] = dA * Sr[n] + du * bcr[n];
            Pr[n] *= dA;
        }
    }
    const size_t base = (((size_t)b * NCHUNK + c) * D_STATE) * D_INNER + d;
#pragma unroll
    for (int n = 0; n < D_STATE; ++n) {
        P[base + (size_t)n * D_INNER] = Pr[n];
        S[base + (size_t)n * D_INNER] = Sr[n];
    }
}

__global__ __launch_bounds__(256)
void k_scan_mid(const float* __restrict__ P, const float* __restrict__ S,
                float* __restrict__ hstart) {
    const int idx = blockIdx.x * 256 + threadIdx.x;   // B*D_INNER threads
    const int d = idx & (D_INNER - 1);
    const int b = idx >> 11;

    float h[D_STATE];
#pragma unroll
    for (int n = 0; n < D_STATE; ++n) h[n] = 0.f;

    for (int c = 0; c < NCHUNK; ++c) {
        const size_t base = (((size_t)b * NCHUNK + c) * D_STATE) * D_INNER + d;
#pragma unroll
        for (int n = 0; n < D_STATE; ++n) {
            const size_t o = base + (size_t)n * D_INNER;
            hstart[o] = h[n];
            h[n] = P[o] * h[n] + S[o];
        }
    }
}

__global__ __launch_bounds__(256)
void k_scan_pass2(const float* __restrict__ delta, const float* __restrict__ u,
                  const float* __restrict__ bc, const float* __restrict__ xz,
                  const float* __restrict__ A_log, const float* __restrict__ Dp,
                  const float* __restrict__ hstart, unsigned short* __restrict__ y_bf) {
    const int idx = blockIdx.x * 256 + threadIdx.x;   // B*NCHUNK*D_INNER threads
    const int d = idx & (D_INNER - 1);
    const int c = (idx >> 11) & (NCHUNK - 1);
    const int b = idx >> 15;

    float a[D_STATE], h[D_STATE];
    const size_t base = (((size_t)b * NCHUNK + c) * D_STATE) * D_INNER + d;
#pragma unroll
    for (int n = 0; n < D_STATE; ++n) {
        a[n] = -__expf(A_log[d * D_STATE + n]);
        h[n] = hstart[base + (size_t)n * D_INNER];
    }
    const float Dv = Dp[d];
    const size_t l0 = (size_t)b * L_SEQ + (size_t)c * CHUNK;

    for (int t = 0; t < CHUNK; ++t) {
        const size_t rl = l0 + t;
        const float dt = delta[rl * D_INNER + d];
        const float uv = u[rl * D_INNER + d];
        const float du = dt * uv;
        const float* bcr = bc + rl * (2 * D_STATE);
        float y = 0.f;
#pragma unroll
        for (int n = 0; n < D_STATE; ++n) {
            const float dA = __expf(dt * a[n]);
            h[n] = dA * h[n] + du * bcr[n];            // B_ssm
            y += h[n] * bcr[D_STATE + n];              // C_ssm
        }
        const float z = xz[rl * (2 * D_INNER) + D_INNER + d];
        const float out = (y + Dv * uv) * silu_(z);
        y_bf[rl * D_INNER + d] = f2bf(out);
    }
}

// ---------------------------------------------------------------------------

extern "C" void kernel_launch(void* const* d_in, const int* in_sizes, int n_in,
                              void* d_out, int out_size, void* d_ws, size_t ws_size,
                              hipStream_t stream) {
    (void)in_sizes; (void)n_in; (void)out_size; (void)ws_size;

    const float* x         = (const float*)d_in[0];
    const float* norm_w    = (const float*)d_in[1];
    const float* norm_b    = (const float*)d_in[2];
    const float* in_proj_w = (const float*)d_in[3];
    const float* in_proj_b = (const float*)d_in[4];
    const float* conv_w    = (const float*)d_in[5];
    const float* conv_b    = (const float*)d_in[6];
    const float* x_proj_w  = (const float*)d_in[7];
    const float* x_proj_b  = (const float*)d_in[8];
    const float* dt_proj_w = (const float*)d_in[9];
    const float* dt_proj_b = (const float*)d_in[10];
    const float* A_log     = (const float*)d_in[11];
    const float* D_param   = (const float*)d_in[12];
    const float* out_proj_w= (const float*)d_in[13];
    const float* out_proj_b= (const float*)d_in[14];
    float* out = (float*)d_out;

    // workspace carve-up (256B aligned)
    char* ws = (char*)d_ws;
    size_t off = 0;
    auto alloc = [&](size_t bytes) { void* p = ws + off; off = (off + bytes + 255) & ~(size_t)255; return p; };

    unsigned short* win_bf  = (unsigned short*)alloc((size_t)2 * D_INNER * D_MODEL * 2); // 4096x1024
    unsigned short* wdt_bf  = (unsigned short*)alloc((size_t)D_INNER * D_INNER * 2);     // 2048x2048
    unsigned short* wout_bf = (unsigned short*)alloc((size_t)D_MODEL * D_INNER * 2);     // 1024x2048
    unsigned short* xn_bf   = (unsigned short*)alloc((size_t)BL * D_MODEL * 2);
    unsigned short* u_bf    = (unsigned short*)alloc((size_t)BL * D_INNER * 2);
    unsigned short* y_bf    = (unsigned short*)alloc((size_t)BL * D_INNER * 2);
    float* xz    = (float*)alloc((size_t)BL * 2 * D_INNER * 4);
    float* u_f   = (float*)alloc((size_t)BL * D_INNER * 4);
    float* deltaf= (float*)alloc((size_t)BL * D_INNER * 4);
    float* bcf   = (float*)alloc((size_t)BL * 2 * D_STATE * 4);
    const size_t scanN = (size_t)B_SZ * NCHUNK * D_STATE * D_INNER;  // 1M floats
    float* Pbuf  = (float*)alloc(scanN * 4);
    float* Sbuf  = (float*)alloc(scanN * 4);
    float* hbuf  = (float*)alloc(scanN * 4);

    // 1) weight conversions
    {
        int n1 = 2 * D_INNER * D_MODEL;
        k_f32_to_bf16<<<(n1 + 255) / 256, 256, 0, stream>>>(in_proj_w, win_bf, n1);
        int n2 = D_INNER * D_INNER;
        k_f32_to_bf16<<<(n2 + 255) / 256, 256, 0, stream>>>(dt_proj_w, wdt_bf, n2);
        int n3 = D_MODEL * D_INNER;
        k_f32_to_bf16<<<(n3 + 255) / 256, 256, 0, stream>>>(out_proj_w, wout_bf, n3);
    }

    // 2) layernorm -> bf16
    k_layernorm_bf16<<<BL, 256, 0, stream>>>(x, norm_w, norm_b, xn_bf);

    // 3) in_proj: xz[2048,4096] = xn[2048,1024] @ win[4096,1024]^T + b
    k_gemm_wmma_bf16<0><<<dim3((2 * D_INNER) / 128, BL / 64), 256, 0, stream>>>(
        xn_bf, win_bf, in_proj_b, nullptr, xz, BL, 2 * D_INNER, D_MODEL);

    // 4) depthwise causal conv + SiLU -> u (f32 + bf16)
    k_conv_silu<<<(BL * D_INNER) / 256, 256, 0, stream>>>(xz, conv_w, conv_b, u_f, u_bf);

    // 5) x_proj -> BC[2048,32]
    k_xproj<<<BL, 256, 0, stream>>>(u_f, x_proj_w, x_proj_b, bcf);

    // 6) dt_proj + softplus: delta[2048,2048]
    k_gemm_wmma_bf16<1><<<dim3(D_INNER / 128, BL / 64), 256, 0, stream>>>(
        u_bf, wdt_bf, dt_proj_b, nullptr, deltaf, BL, D_INNER, D_INNER);

    // 7) chunked selective scan (3 phases) -> y (bf16), fused +u*D and *silu(z)
    k_scan_pass1<<<(B_SZ * NCHUNK * D_INNER) / 256, 256, 0, stream>>>(
        deltaf, u_f, bcf, A_log, Pbuf, Sbuf);
    k_scan_mid<<<(B_SZ * D_INNER) / 256, 256, 0, stream>>>(Pbuf, Sbuf, hbuf);
    k_scan_pass2<<<(B_SZ * NCHUNK * D_INNER) / 256, 256, 0, stream>>>(
        deltaf, u_f, bcf, xz, A_log, D_param, hbuf, y_bf);

    // 8) out_proj + residual -> d_out
    k_gemm_wmma_bf16<2><<<dim3(D_MODEL / 128, BL / 64), 256, 0, stream>>>(
        y_bf, wout_bf, out_proj_b, x, out, BL, D_MODEL, D_INNER);
}
